// OffsetModel_16518444220826
// MI455X (gfx1250) — matrix operand order repaired and checked by reading, not compiled, not run on hardware
//
#include <hip/hip_runtime.h>
#include <hip/hip_bf16.h>
#include <stdint.h>

typedef __bf16 bf16_t;
typedef __attribute__((ext_vector_type(16))) __bf16 v16bf;
typedef __attribute__((ext_vector_type(8)))  __bf16 v8bf;
typedef __attribute__((ext_vector_type(8)))  float  v8f;

#define NPTS  300000
#define CH    128
#define KNB   27
#define MTILE 128   // rows per block
#define TPB   256   // 8 waves (wave32)

// ---------------------------------------------------------------------------
// CDNA5 async global->LDS copy (ASYNCcnt-tracked, no VGPR round trip).
// lds = low 32 bits of generic shared address (== LDS byte offset),
// g   = 64-bit global address, OFF = immediate byte offset.
// ---------------------------------------------------------------------------
template <int OFF>
__device__ __forceinline__ void async_b128(uint32_t lds, uint64_t g) {
    asm volatile("global_load_async_to_lds_b128 %0, %1, off offset:%2"
                 :: "v"(lds), "v"(g), "i"(OFF) : "memory");
}
__device__ __forceinline__ void copy128_async(uint32_t lds, const void* gp) {
    uint64_t g = (uint64_t)(uintptr_t)gp;
    async_b128<0>(lds, g);   async_b128<16>(lds, g);
    async_b128<32>(lds, g);  async_b128<48>(lds, g);
    async_b128<64>(lds, g);  async_b128<80>(lds, g);
    async_b128<96>(lds, g);  async_b128<112>(lds, g);
}
__device__ __forceinline__ void wait_async0() {
    asm volatile("s_wait_asynccnt 0x0" ::: "memory");
}

// ---------------------------------------------------------------------------
// Zero a small region (zero page for invalid-neighbor gathers).
// ---------------------------------------------------------------------------
__global__ void k_zero(uint32_t* p, int n) {
    int i = blockIdx.x * blockDim.x + threadIdx.x;
    if (i < n) p[i] = 0u;
}

// ---------------------------------------------------------------------------
// Convert f32 weights [nmat][Cin=128][Cout=128] -> bf16 transposed
// [nmat][Cout=128][Cin=128] so WMMA B-fragments read contiguous Cin.
// ---------------------------------------------------------------------------
__global__ void k_wconv(const float* __restrict__ src, bf16_t* __restrict__ dst,
                        int nmat) {
    long i = (long)blockIdx.x * blockDim.x + threadIdx.x;
    long total = (long)nmat * CH * CH;
    if (i >= total) return;
    int ci = (int)(i % CH);
    int co = (int)((i / CH) % CH);
    long m = i / ((long)CH * CH);
    dst[i] = (bf16_t)src[(m * CH + ci) * CH + co];
}

// ---------------------------------------------------------------------------
// Input layer: h = relu( sum_k x_feat[nbr[k]] * W_in[k] + b_in ), bf16 out.
// ---------------------------------------------------------------------------
__global__ void k_input(const float* __restrict__ xf, const int* __restrict__ nbr,
                        const float* __restrict__ Win, const float* __restrict__ bin,
                        bf16_t* __restrict__ h) {
    long row = blockIdx.x;
    int  c   = threadIdx.x;
    float acc = bin[c];
    #pragma unroll 1
    for (int k = 0; k < KNB; ++k) {
        int idx = nbr[row * KNB + k];
        if (idx >= 0) acc += xf[idx] * Win[k * CH + c];
    }
    h[row * CH + c] = (bf16_t)fmaxf(acc, 0.f);
}

// ---------------------------------------------------------------------------
// Fused gather-GEMM sconv, bf16 WMMA -> f32 acc, async double-buffered LDS.
//   dst[n, :] = maybe_relu( sum_k src[nbr[n,k], :] @ W[k] + bias (+ addsrc) )
// nbr == nullptr -> identity gather (dense GEMM), nK = 1.
// wT layout: [nK][Cout=128][Cin=128] bf16.
// Waves arranged 4(M) x 2(N); each wave owns a 32x64 output tile.
// ---------------------------------------------------------------------------
__global__ __launch_bounds__(TPB)
void k_sconv(const bf16_t* __restrict__ src, const int* __restrict__ nbr, int nK,
             const bf16_t* __restrict__ wT, const float* __restrict__ bias,
             const bf16_t* __restrict__ addsrc, bf16_t* __restrict__ dst,
             int do_relu, const bf16_t* __restrict__ zpage) {
    __shared__ __align__(32) bf16_t sA[2][MTILE * CH];  // 2 x 32 KB activations
    __shared__ __align__(32) bf16_t sW[2][CH * CH];     // 2 x 32 KB weights

    const int  tid  = threadIdx.x;
    const int  wave = tid >> 5;
    const int  lane = tid & 31;
    const int  ln15 = lane & 15;
    const bool lo   = lane < 16;
    const int  mw   = wave >> 1;   // 0..3 -> rows 32*mw
    const int  nw   = wave & 1;    // 0..1 -> cols 64*nw
    const long row0 = (long)blockIdx.x * MTILE;

    const int  r_half = tid >> 1;  // row handled by this thread in staging
    const int  half   = tid & 1;   // which 64-channel half
    const long growS  = row0 + r_half;

    auto stage = [&](int k, int buf) {
        // gathered activation rows (128B per thread, branch-free via zero page)
        const bf16_t* s = zpage;
        if (growS < NPTS) {
            long idx = nbr ? (long)nbr[growS * KNB + k] : growS;
            if (idx >= 0) s = src + idx * CH + half * 64;
        }
        copy128_async((uint32_t)(uintptr_t)&sA[buf][r_half * CH + half * 64], s);
        // weight tile W[k], contiguous 32 KB
        copy128_async((uint32_t)(uintptr_t)&sW[buf][tid * 64],
                      (const char*)(wT + (long)k * CH * CH) + tid * 128);
    };

    const v8f vz = {0.f, 0.f, 0.f, 0.f, 0.f, 0.f, 0.f, 0.f};
    v8f acc[8];
    #pragma unroll
    for (int t = 0; t < 8; ++t) acc[t] = vz;

    stage(0, 0);
    for (int k = 0; k < nK; ++k) {
        const int cur = k & 1;
        wait_async0();      // my stage(k) ops complete
        __syncthreads();    // everyone's stage(k) complete
        if (k + 1 < nK) stage(k + 1, cur ^ 1);   // overlaps with compute below

        const char* pA = (const char*)&sA[cur][0];
        const char* pW = (const char*)&sW[cur][0];
        #pragma unroll
        for (int ks = 0; ks < 4; ++ks) {
            // A fragments (16x32 bf16, contiguous layout: lanes 0-15 K{0..15}, lanes 16-31 K{16..31})
            union { v16bf v; v8bf h[2]; } a[2];
            #pragma unroll
            for (int mi = 0; mi < 2; ++mi) {
                int ab = (mw * 32 + mi * 16 + ln15) * 256 + ks * 64 + (lo ? 0 : 32);
                a[mi].h[0] = *(const v8bf*)(pA + ab);
                a[mi].h[1] = *(const v8bf*)(pA + ab + 16);
            }
            #pragma unroll
            for (int t = 0; t < 4; ++t) {
                // B fragment (32x16): lane n reads 16 contiguous Cin values
                int bb = (nw * 64 + t * 16 + ln15) * 256 + ks * 64 + (lo ? 0 : 32);
                v16bf b = *(const v16bf*)(pW + bb);
                #pragma unroll
                for (int mi = 0; mi < 2; ++mi)
                    acc[mi * 4 + t] = __builtin_amdgcn_wmma_f32_16x16x32_bf16(
                        false, a[mi].v, false, b, (short)0, acc[mi * 4 + t],
                        false, false);
            }
        }
        __syncthreads();    // done reading sA/sW[cur] before it is restaged
    }

    // ---- epilogue: bias into LDS (C/D layout), then coalesced fused store ----
    bf16_t* sOut = &sA[0][0];
    #pragma unroll
    for (int mi = 0; mi < 2; ++mi) {
        #pragma unroll
        for (int t = 0; t < 4; ++t) {
            int   col = nw * 64 + t * 16 + ln15;
            float bv  = bias[col];
            #pragma unroll
            for (int r = 0; r < 8; ++r) {
                int lrow = mw * 32 + mi * 16 + r + (lo ? 0 : 8);
                sOut[lrow * CH + col] = (bf16_t)(acc[mi * 4 + t][r] + bv);
            }
        }
    }
    __syncthreads();
    if (growS < NPTS) {
        const int cbase = half * 64;
        #pragma unroll
        for (int j = 0; j < 8; ++j) {
            union { v8bf v; bf16_t e[8]; } xa, xb, xo;
            xa.v = *(const v8bf*)&sOut[r_half * CH + cbase + j * 8];
            if (addsrc)
                xb.v = *(const v8bf*)(addsrc + growS * CH + cbase + j * 8);
            #pragma unroll
            for (int e = 0; e < 8; ++e) {
                float v = (float)xa.e[e];
                if (addsrc)  v += (float)xb.e[e];
                if (do_relu) v  = fmaxf(v, 0.f);
                xo.e[e] = (bf16_t)v;
            }
            *(v8bf*)(dst + growS * CH + cbase + j * 8) = xo.v;
        }
    }
}

// ---------------------------------------------------------------------------
// Final 128 -> 3 layer, f32 output.
// ---------------------------------------------------------------------------
__global__ void k_final(const bf16_t* __restrict__ h, const float* __restrict__ W,
                        const float* __restrict__ b, float* __restrict__ out) {
    long row = (long)blockIdx.x * blockDim.x + threadIdx.x;
    if (row >= NPTS) return;
    float a0 = b[0], a1 = b[1], a2 = b[2];
    const bf16_t* hr = h + row * CH;
    #pragma unroll 4
    for (int c = 0; c < CH; ++c) {
        float x = (float)hr[c];
        a0 += x * W[c * 3 + 0];
        a1 += x * W[c * 3 + 1];
        a2 += x * W[c * 3 + 2];
    }
    out[row * 3 + 0] = a0;
    out[row * 3 + 1] = a1;
    out[row * 3 + 2] = a2;
}

// ---------------------------------------------------------------------------
extern "C" void kernel_launch(void* const* d_in, const int* in_sizes, int n_in,
                              void* d_out, int out_size, void* d_ws, size_t ws_size,
                              hipStream_t stream) {
    const float* x_feat = (const float*)d_in[0];
    const float* W_in   = (const float*)d_in[1];
    const float* b_in   = (const float*)d_in[2];
    const float* W_res  = (const float*)d_in[3];
    const float* b_res  = (const float*)d_in[4];
    const float* W_out  = (const float*)d_in[5];
    const float* b_out  = (const float*)d_in[6];
    const float* Wf1    = (const float*)d_in[7];
    const float* bf1    = (const float*)d_in[8];
    const float* Wf2    = (const float*)d_in[9];
    const float* bf2    = (const float*)d_in[10];
    const float* Wf3    = (const float*)d_in[11];
    const float* bf3    = (const float*)d_in[12];
    const int*   nbr    = (const int*)d_in[13];
    float*       out    = (float*)d_out;
    (void)in_sizes; (void)n_in; (void)out_size; (void)ws_size;

    char*  ws  = (char*)d_ws;
    size_t off = 0;
    auto take = [&](size_t bytes) {
        char* p = ws + off;
        off = (off + bytes + 255) & ~(size_t)255;
        return p;
    };
    bf16_t* hbuf   = (bf16_t*)take((size_t)NPTS * CH * 2);
    bf16_t* rbuf   = (bf16_t*)take((size_t)NPTS * CH * 2);
    bf16_t* wres_t = (bf16_t*)take((size_t)6 * KNB * CH * CH * 2);
    bf16_t* wout_t = (bf16_t*)take((size_t)KNB * CH * CH * 2);
    bf16_t* wf1_t  = (bf16_t*)take((size_t)CH * CH * 2);
    bf16_t* wf2_t  = (bf16_t*)take((size_t)CH * CH * 2);
    bf16_t* zpage  = (bf16_t*)take((size_t)256);

    // zero page + one-time weight convert/transpose to bf16 [k][Cout][Cin]
    k_zero<<<1, 64, 0, stream>>>((uint32_t*)zpage, 64);
    {
        long tot = (long)6 * KNB * CH * CH;
        k_wconv<<<(unsigned)((tot + 255) / 256), 256, 0, stream>>>(W_res, wres_t, 6 * KNB);
        tot = (long)KNB * CH * CH;
        k_wconv<<<(unsigned)((tot + 255) / 256), 256, 0, stream>>>(W_out, wout_t, KNB);
        k_wconv<<<(CH * CH + 255) / 256, 256, 0, stream>>>(Wf1, wf1_t, 1);
        k_wconv<<<(CH * CH + 255) / 256, 256, 0, stream>>>(Wf2, wf2_t, 1);
    }

    // h0 = relu(sconv(x_feat, W_in) + b_in)
    k_input<<<NPTS, CH, 0, stream>>>(x_feat, nbr, W_in, b_in, hbuf);

    dim3 g((NPTS + MTILE - 1) / MTILE), blk(TPB);
    for (int i = 0; i < 3; ++i) {
        // r = relu(sconv(h, W_res[i,0]) + b)
        k_sconv<<<g, blk, 0, stream>>>(hbuf, nbr, KNB,
                                       wres_t + (size_t)(i * 2 + 0) * KNB * CH * CH,
                                       b_res + (i * 2 + 0) * CH, nullptr, rbuf, 1, zpage);
        // h = h + sconv(r, W_res[i,1]) + b
        k_sconv<<<g, blk, 0, stream>>>(rbuf, nbr, KNB,
                                       wres_t + (size_t)(i * 2 + 1) * KNB * CH * CH,
                                       b_res + (i * 2 + 1) * CH, hbuf, hbuf, 0, zpage);
    }
    // h = sconv(h, W_out) + b_out
    k_sconv<<<g, blk, 0, stream>>>(hbuf, nbr, KNB, wout_t, b_out, nullptr, rbuf, 0, zpage);
    // h = relu(h @ Wf1 + bf1)   (dense: identity gather)
    k_sconv<<<g, blk, 0, stream>>>(rbuf, nullptr, 1, wf1_t, bf1, nullptr, hbuf, 1, zpage);
    // h = relu(h @ Wf2 + bf2)
    k_sconv<<<g, blk, 0, stream>>>(hbuf, nullptr, 1, wf2_t, bf2, nullptr, rbuf, 1, zpage);
    // out = h @ Wf3 + bf3
    k_final<<<(NPTS + 255) / 256, 256, 0, stream>>>(rbuf, Wf3, bf3, out);
}
